// StrideHybridModel_11570641895810
// MI455X (gfx1250) — compile-verified, matrix operand order checked by reading, not run on hardware
//
#include <hip/hip_runtime.h>
#include <hip/hip_bf16.h>

// ---------------------------------------------------------------------------
// StrideHybridModel forward for MI455X (gfx1250, wave32, WMMA).
//
// Chunked-parallel schedule: SSM slot only changes at t=16k, so windows are
// independent inside chunks [0,16], [16k+1,16k+16]. All GEMMs run as f16
// v_wmma_f32_16x16x32_f16 with f32 accumulation.
//
// B operands (weights, L2-resident) are pre-packed fragment-major so each lane
// fragment is one 32B contiguous chunk (2x global_load_b128). A tiles stream
// through a double-buffered LDS stage via GLOBAL_LOAD_ASYNC_TO_LDS_B128
// (ASYNCcnt), with one barrier per K-step. M = W*80 is a multiple of 64, so
// main-GEMM epilogues and A loads are guard-free.
// ---------------------------------------------------------------------------

typedef _Float16 h16;
typedef __attribute__((ext_vector_type(16))) _Float16 v16h;
typedef __attribute__((ext_vector_type(8)))  float    v8f;
typedef __attribute__((__vector_size__(16))) int      v4i_t;
typedef __attribute__((address_space(1))) v4i_t       as1_v4i;
typedef __attribute__((address_space(3))) v4i_t       as3_v4i;

#define DM     512
#define NHEAD  8
#define HD     64
#define DFFN   2048
#define TWIN   65
#define RPAD   80      // window rows padded to 5x16 tiles
#define VOCAB  32000
#define BATCH  4
#define SEQ    256
#define WMAX   68      // max windows per chunk (chunk 0: 17 steps * 4 batch)
#define TROWS  128     // tlast rows (2 M-blocks, all defined)

#define LDA_P  40      // LDS A tile stride (32+8 halves)

#if __has_builtin(__builtin_amdgcn_global_load_async_to_lds_b128)
#define ASYNC_A 1
#else
#define ASYNC_A 0
#endif

__device__ __forceinline__ void a_stage16(const h16* __restrict__ g, h16* __restrict__ l) {
#if ASYNC_A
  __builtin_amdgcn_global_load_async_to_lds_b128((as1_v4i*)g, (as3_v4i*)l, 0, 0);
#else
  float4 r = *(const float4*)g;
  *(float4*)l = r;
#endif
}

__device__ __forceinline__ void a_stage_wait() {
#if ASYNC_A
#if __has_builtin(__builtin_amdgcn_s_wait_asynccnt)
  __builtin_amdgcn_s_wait_asynccnt(0);
#else
  asm volatile("s_wait_asynccnt 0x0" ::: "memory");
#endif
#endif
}

// ------------------------------ utility kernels -----------------------------

__global__ void zero_f32(float* __restrict__ p, long n) {
  long i = (long)blockIdx.x * blockDim.x + threadIdx.x;
  long st = (long)gridDim.x * blockDim.x;
  for (; i < n; i += st) p[i] = 0.f;
}

// Pack a f32 weight matrix B[K][N] (or, if trans, B[k][n] = src[n*K + k]) into
// f16 WMMA-fragment-major order:
//   dst[ ((k/32)*(N/16) + n/16)*512 + lane*16 + j ]
// with lane = 16*((k%32)>>3 & 1) + (n%16),  j = (k%8) | ((k%32)&16 ? 8 : 0),
// matching the 16-bit B VGPR layout of V_WMMA_F32_16X16X32_F16.
__global__ void pack_b_f32(const float* __restrict__ src, h16* __restrict__ dst,
                           int K, int N, int trans) {
  long total = (long)K * N;
  long i = (long)blockIdx.x * blockDim.x + threadIdx.x;
  long st = (long)gridDim.x * blockDim.x;
  for (; i < total; i += st) {
    long k = i / N, n = i - k * N;
    float v = trans ? src[n * (long)K + k] : src[k * (long)N + n];
    int kloc = (int)(k & 31);
    int lh = (kloc >> 3) & 1;
    int j = (kloc & 7) | ((kloc & 16) ? 8 : 0);
    long idx = (((k >> 5) * (N >> 4) + (n >> 4)) << 9) + (long)(lh * 16 + (n & 15)) * 16 + j;
    dst[idx] = (h16)v;
  }
}

// ------------------------------ window builder ------------------------------
__global__ void build_window(const int* __restrict__ ids, const float* __restrict__ emb,
                             const float* __restrict__ ssm_row, float* __restrict__ xbuf,
                             int lo) {
  int w = blockIdx.x;
  int b = w & 3, t = lo + (w >> 2);
  float* xw = xbuf + (long)w * RPAD * DM;
  for (int idx = threadIdx.x; idx < RPAD * DM; idx += blockDim.x) {
    int r = idx >> 9, c = idx & 511;
    float v = 0.f;
    if (r == 0) v = ssm_row[b * DM + c];
    else if (r <= 64) {
      int tok = t - 64 + r;
      if (tok >= 0) v = emb[(long)ids[b * SEQ + tok] * DM + c];
    }
    xw[idx] = v;
  }
}

// ------------------------------ layernorm -----------------------------------
__global__ __launch_bounds__(128)
void ln_rows(const float* __restrict__ x, h16* __restrict__ y,
             const float* __restrict__ g, const float* __restrict__ bb) {
  int row = blockIdx.x;
  int sub = row % RPAD;
  const float* xr = x + (long)row * DM;
  h16* yr = y + (long)row * DM;
  int tid = threadIdx.x;
  if (sub >= TWIN) { for (int c = tid; c < DM; c += 128) yr[c] = (h16)0.f; return; }
  __shared__ float s1[128], s2[128];
  float a = 0.f, q = 0.f;
  for (int c = tid; c < DM; c += 128) { float v = xr[c]; a += v; q += v * v; }
  s1[tid] = a; s2[tid] = q; __syncthreads();
  for (int off = 64; off > 0; off >>= 1) {
    if (tid < off) { s1[tid] += s1[tid + off]; s2[tid] += s2[tid + off]; }
    __syncthreads();
  }
  float mu = s1[0] * (1.f / DM);
  float var = s2[0] * (1.f / DM) - mu * mu;
  float rs = rsqrtf(var + 1e-5f);
  for (int c = tid; c < DM; c += 128) yr[c] = (h16)((xr[c] - mu) * rs * g[c] + bb[c]);
}

// final LN of last window row -> tlast[128][512] f16 (rows >= W zeroed so the
// logits GEMM can load A unconditionally)
__global__ __launch_bounds__(128)
void lnf_last(const float* __restrict__ xbuf, h16* __restrict__ tlast,
              const float* __restrict__ g, const float* __restrict__ bb, int W) {
  int w = blockIdx.x, tid = threadIdx.x;
  h16* yr = tlast + (long)w * DM;
  if (w >= W) { for (int c = tid; c < DM; c += 128) yr[c] = (h16)0.f; return; }
  const float* xr = xbuf + ((long)w * RPAD + 64) * DM;
  __shared__ float s1[128], s2[128];
  float a = 0.f, q = 0.f;
  for (int c = tid; c < DM; c += 128) { float v = xr[c]; a += v; q += v * v; }
  s1[tid] = a; s2[tid] = q; __syncthreads();
  for (int off = 64; off > 0; off >>= 1) {
    if (tid < off) { s1[tid] += s1[tid + off]; s2[tid] += s2[tid + off]; }
    __syncthreads();
  }
  float mu = s1[0] * (1.f / DM);
  float var = s2[0] * (1.f / DM) - mu * mu;
  float rs = rsqrtf(var + 1e-5f);
  for (int c = tid; c < DM; c += 128) yr[c] = (h16)((xr[c] - mu) * rs * g[c] + bb[c]);
}

// ------------------------------ WMMA GEMM -----------------------------------
// Block tile 64x256, 8 waves in 2x4 grid, wave tile 32x64 (8 v_wmma / K-step).
// MODE: 0 store f16 (QKV) | 1 bias+GELU f16 (FFN1) | 2 [+bias] += f32 residual
// (WO/FFN2) | 3 scatter f32 logits rows to d_out.
template<int MODE>
__global__ __launch_bounds__(256)
void gemm_wmma(const h16* __restrict__ A, const h16* __restrict__ Bp,
               int M, int K, int N, int Mstore,
               const float* __restrict__ bias,
               float* __restrict__ Cf, h16* __restrict__ Ch,
               int lo, float* __restrict__ outL) {
  __shared__ h16 As[2][64 * LDA_P];
  int tid = threadIdx.x;
  int lane = tid & 31, wid = tid >> 5;
  int wm = wid >> 2, wn = wid & 3;          // 2x4 wave grid
  int bm = blockIdx.y * 64, bn = blockIdx.x * 256;
  int l15 = lane & 15, lh = lane >> 4, kb = lh * 8;
  int ntN = N >> 4;
  int nt0 = (bn >> 4) + wn * 4;

  v8f acc[2][4];
  #pragma unroll
  for (int i = 0; i < 2; ++i)
    #pragma unroll
    for (int j = 0; j < 4; ++j)
      #pragma unroll
      for (int r = 0; r < 8; ++r) acc[i][j][r] = 0.f;

  // A-tile fill mapping: 64 rows x 32 halves, 16B per thread (always in-bounds:
  // M is a multiple of 64 for modes 0-2; tlast has 128 defined rows for mode 3)
  int ar = tid >> 2, ac = (tid & 3) * 8;
  const h16* aglob = A + (long)(bm + ar) * K + ac;
  h16* lds0 = &As[0][ar * LDA_P + ac];
  h16* lds1 = &As[1][ar * LDA_P + ac];
  const h16* bpl = Bp + (long)lane * 16;

  int NK = K >> 5;
  a_stage16(aglob, lds0);                   // preload tile 0
  a_stage_wait();
  __syncthreads();

  for (int it = 0; it < NK; ++it) {
    int cur = it & 1;
    if (it + 1 < NK)                        // prefetch next tile -> other buffer
      a_stage16(aglob + (long)(it + 1) * 32, (it & 1) ? lds0 : lds1);

    v16h af[2];
    #pragma unroll
    for (int i = 0; i < 2; ++i) {           // A frags: ds_load_b128 x2
      const h16* ap = &As[cur][(wm * 32 + i * 16 + l15) * LDA_P];
      #pragma unroll
      for (int j = 0; j < 8; ++j) {
        int kx = kb + 2 * j + ((j >= 4) ? 8 : 0);
        af[i][2 * j]     = ap[kx];
        af[i][2 * j + 1] = ap[kx + 1];
      }
    }
    v16h bf[4];
    long kbase = (long)it * ntN;
    #pragma unroll
    for (int jn = 0; jn < 4; ++jn)          // B frags: 32B contiguous per lane
      bf[jn] = *(const v16h*)(bpl + ((kbase + nt0 + jn) << 9));
    #pragma unroll
    for (int i = 0; i < 2; ++i)
      #pragma unroll
      for (int jn = 0; jn < 4; ++jn)
        acc[i][jn] = __builtin_amdgcn_wmma_f32_16x16x32_f16(
            false, af[i], false, bf[jn], (short)0, acc[i][jn], false, false);

    a_stage_wait();
    __syncthreads();
  }

  // epilogue (C layout: lane = col, VGPR r = row r + 8*laneHalf)
  #pragma unroll
  for (int i = 0; i < 2; ++i) {
    #pragma unroll
    for (int jn = 0; jn < 4; ++jn) {
      int gn = bn + wn * 64 + jn * 16 + l15;
      int gm0 = bm + wm * 32 + i * 16 + lh * 8;
      if (MODE == 0) {
        h16* p = Ch + (long)gm0 * N + gn;
        #pragma unroll
        for (int r = 0; r < 8; ++r) { *p = (h16)acc[i][jn][r]; p += N; }
      } else if (MODE == 1) {
        float bv = bias[gn];
        h16* p = Ch + (long)gm0 * N + gn;
        #pragma unroll
        for (int r = 0; r < 8; ++r) {
          float v = acc[i][jn][r] + bv;
          float t = 0.7978845608028654f * (v + 0.044715f * v * v * v);
          *p = (h16)(0.5f * v * (1.f + tanhf(t)));
          p += N;
        }
      } else if (MODE == 2) {
        float bv = bias ? bias[gn] : 0.f;
        float* p = Cf + (long)gm0 * N + gn;
        #pragma unroll
        for (int r = 0; r < 8; ++r) { *p += acc[i][jn][r] + bv; p += N; }
      } else {
        #pragma unroll
        for (int r = 0; r < 8; ++r) {
          int gm = gm0 + r;
          if (gm < Mstore)
            outL[((long)((gm & 3) * SEQ + lo + (gm >> 2))) * VOCAB + gn] = acc[i][jn][r];
        }
      }
    }
  }
}

// ------------------------------ attention -----------------------------------
__global__ __launch_bounds__(256)
void attn_kernel(const h16* __restrict__ qkv, h16* __restrict__ obuf) {
  __shared__ h16 Ks[TWIN * HD];
  __shared__ h16 Vs[TWIN * HD];
  __shared__ float pb[8][WMAX];
  int w = blockIdx.x >> 3, hh = blockIdx.x & 7;
  int tid = threadIdx.x, lane = tid & 31, wid = tid >> 5;
  const h16* base = qkv + (long)w * RPAD * (3 * DM);

  for (int idx = tid; idx < TWIN * HD; idx += 256) {
    int r = idx >> 6, d = idx & 63;
    Ks[idx] = base[r * (3 * DM) + DM + hh * HD + d];
    Vs[idx] = base[r * (3 * DM) + 2 * DM + hh * HD + d];
  }
  for (int idx = tid; idx < 15 * HD; idx += 256) {   // zero padded output rows
    int r = TWIN + (idx >> 6), d = idx & 63;
    obuf[((long)w * RPAD + r) * DM + hh * HD + d] = (h16)0.f;
  }
  __syncthreads();

  for (int i = wid; i < TWIN; i += 8) {
    const h16* qrow = base + (long)i * (3 * DM) + hh * HD;
    float qa = (float)qrow[lane];
    float qb = (float)qrow[lane + 32];
    float s0 = 0.f, s1 = 0.f, s2 = 0.f;       // scores for j=lane, lane+32, 64
    #pragma unroll 8
    for (int d = 0; d < 32; ++d) {
      float q0 = __shfl(qa, d);
      float q1 = __shfl(qb, d);
      s0 += q0 * (float)Ks[lane * HD + d]        + q1 * (float)Ks[lane * HD + d + 32];
      s1 += q0 * (float)Ks[(lane + 32) * HD + d] + q1 * (float)Ks[(lane + 32) * HD + d + 32];
      s2 += q0 * (float)Ks[64 * HD + d]          + q1 * (float)Ks[64 * HD + d + 32];
    }
    s0 *= 0.125f; s1 *= 0.125f; s2 *= 0.125f;
    bool v0 = (lane <= i), v1 = (lane + 32 <= i), v2 = (i == 64);
    const float NEG = -1e30f;
    float m = v0 ? s0 : NEG;
    m = fmaxf(m, v1 ? s1 : NEG);
    m = fmaxf(m, v2 ? s2 : NEG);
    for (int off = 16; off; off >>= 1) m = fmaxf(m, __shfl_xor(m, off));
    float p0 = v0 ? __expf(s0 - m) : 0.f;
    float p1 = v1 ? __expf(s1 - m) : 0.f;
    float p2 = v2 ? __expf(s2 - m) : 0.f;
    float sum = p0 + p1 + ((lane == 0) ? p2 : 0.f);
    for (int off = 16; off; off >>= 1) sum += __shfl_xor(sum, off);
    float inv = 1.f / sum;
    pb[wid][lane] = p0 * inv;
    pb[wid][lane + 32] = p1 * inv;
    if (lane == 0) pb[wid][64] = p2 * inv;    // same-wave LDS: in-order per wave
    float o0 = 0.f, o1 = 0.f;
    for (int j = 0; j <= i; ++j) {
      float p = pb[wid][j];
      o0 += p * (float)Vs[j * HD + lane];
      o1 += p * (float)Vs[j * HD + lane + 32];
    }
    h16* orow = obuf + ((long)w * RPAD + i) * DM + hh * HD;
    orow[lane] = (h16)o0;
    orow[lane + 32] = (h16)o1;
  }
}

// ------------------------------ pool + SSM ----------------------------------
__global__ __launch_bounds__(256)
void pool_ssm_update(const int* __restrict__ ids, const float* __restrict__ emb,
                     float* __restrict__ ssm_row, float* __restrict__ hstate,
                     float* __restrict__ wbuf,
                     const float* __restrict__ pool_q, const float* __restrict__ pool_wk,
                     const float* __restrict__ ssm_win, const float* __restrict__ ssm_wdt,
                     const float* __restrict__ ssm_bdt, const float* __restrict__ ssm_Alog,
                     const float* __restrict__ ssm_wB, const float* __restrict__ ssm_wC,
                     const float* __restrict__ ssm_Dw, const float* __restrict__ ssm_wout,
                     int tw) {
  int b = blockIdx.x, tid = threadIdx.x;
  __shared__ float xcur[DM];
  __shared__ float uvec[DM];
  __shared__ float scw[TWIN];
  __shared__ float x1[1024], zt[1024], dtv[1024], yv[1024];
  __shared__ float Bm[16], Cm[16];
  float* wb = wbuf + (long)b * TWIN * DM;

  for (int idx = tid; idx < TWIN * DM; idx += 256) {   // rebuild raw window
    int r = idx >> 9, c = idx & 511;
    float v;
    if (r == 0) v = ssm_row[b * DM + c];
    else { int tok = tw - 64 + r; v = (tok >= 0) ? emb[(long)ids[b * SEQ + tok] * DM + c] : 0.f; }
    wb[idx] = v;
  }
  for (int c = tid; c < DM; c += 256) {                // u = wk @ q
    float s = 0.f;
    for (int d = 0; d < 64; ++d) s += pool_wk[c * 64 + d] * pool_q[d];
    uvec[c] = s;
  }
  __syncthreads();
  if (tid < TWIN) {
    float s = 0.f;
    for (int c = 0; c < DM; ++c) s += wb[tid * DM + c] * uvec[c];
    scw[tid] = s * 0.125f;
  }
  __syncthreads();
  if (tid == 0) {
    float m = scw[0];
    for (int r = 1; r < TWIN; ++r) m = fmaxf(m, scw[r]);
    float s = 0.f;
    for (int r = 0; r < TWIN; ++r) { scw[r] = __expf(scw[r] - m); s += scw[r]; }
    float inv = 1.f / s;
    for (int r = 0; r < TWIN; ++r) scw[r] *= inv;
  }
  __syncthreads();
  for (int c = tid; c < DM; c += 256) {
    float s = 0.f;
    for (int r = 0; r < TWIN; ++r) s += scw[r] * wb[r * DM + c];
    xcur[c] = s;
  }
  __syncthreads();

  for (int l = 0; l < 2; ++l) {
    const float* win  = ssm_win  + (long)l * DM * 2048;
    const float* wdt  = ssm_wdt  + (long)l * 1024 * 1024;
    const float* bdt  = ssm_bdt  + l * 1024;
    const float* Alog = ssm_Alog + l * 1024 * 16;
    const float* wB   = ssm_wB   + l * 1024 * 16;
    const float* wC   = ssm_wC   + l * 1024 * 16;
    const float* Dw   = ssm_Dw   + l * 1024;
    const float* wout = ssm_wout + (long)l * 1024 * DM;

    for (int k = 0; k < 4; ++k) {
      int d = tid + k * 256;
      float a = 0.f, z = 0.f;
      for (int c = 0; c < DM; ++c) {
        float xc = xcur[c];
        a += xc * win[c * 2048 + d];
        z += xc * win[c * 2048 + 1024 + d];
      }
      x1[d] = a; zt[d] = z;
    }
    __syncthreads();
    for (int k = 0; k < 4; ++k) {
      int d = tid + k * 256;
      float a = bdt[d];
      for (int e = 0; e < 1024; ++e) a += x1[e] * wdt[e * 1024 + d];
      dtv[d] = (a > 20.f) ? a : log1pf(__expf(a));     // softplus
    }
    if (tid < 16) {
      float a = 0.f;
      for (int e = 0; e < 1024; ++e) a += x1[e] * wB[e * 16 + tid];
      Bm[tid] = a;
    } else if (tid < 32) {
      int s = tid - 16; float a = 0.f;
      for (int e = 0; e < 1024; ++e) a += x1[e] * wC[e * 16 + s];
      Cm[s] = a;
    }
    __syncthreads();
    for (int k = 0; k < 4; ++k) {
      int d = tid + k * 256;
      float dt = dtv[d], xv = x1[d], y = 0.f;
      long hb = (((long)l * BATCH + b) * 1024 + d) * 16;
      for (int s = 0; s < 16; ++s) {
        float A = -__expf(Alog[d * 16 + s]);
        float hv = __expf(dt * A) * hstate[hb + s] + dt * Bm[s] * xv;
        hstate[hb + s] = hv;
        y += hv * Cm[s];
      }
      y += Dw[d] * xv;
      float zv = zt[d];
      y *= zv / (1.f + __expf(-zv));                   // * silu(z)
      yv[d] = y;
    }
    __syncthreads();
    for (int c = tid; c < DM; c += 256) {
      float s = xcur[c];
      for (int d = 0; d < 1024; ++d) s += yv[d] * wout[d * DM + c];
      xcur[c] = s;
    }
    __syncthreads();
  }
  for (int c = tid; c < DM; c += 256) ssm_row[b * DM + c] = xcur[c];
}

// ------------------------------ host orchestration --------------------------

extern "C" void kernel_launch(void* const* d_in, const int* in_sizes, int n_in,
                              void* d_out, int out_size, void* d_ws, size_t ws_size,
                              hipStream_t stream) {
  const int*   ids     = (const int*)d_in[0];
  const float* emb     = (const float*)d_in[1];
  const float* ln1_g   = (const float*)d_in[2];
  const float* ln1_b   = (const float*)d_in[3];
  const float* wqkv    = (const float*)d_in[4];
  const float* wo      = (const float*)d_in[5];
  const float* ln2_g   = (const float*)d_in[6];
  const float* ln2_b   = (const float*)d_in[7];
  const float* w1      = (const float*)d_in[8];
  const float* b1      = (const float*)d_in[9];
  const float* w2      = (const float*)d_in[10];
  const float* b2      = (const float*)d_in[11];
  const float* lnf_g   = (const float*)d_in[12];
  const float* lnf_b   = (const float*)d_in[13];
  const float* pool_q  = (const float*)d_in[14];
  const float* pool_wk = (const float*)d_in[15];
  const float* ssm_win = (const float*)d_in[16];
  const float* ssm_wdt = (const float*)d_in[17];
  const float* ssm_bdt = (const float*)d_in[18];
  const float* ssm_Alog= (const float*)d_in[19];
  const float* ssm_wB  = (const float*)d_in[20];
  const float* ssm_wC  = (const float*)d_in[21];
  const float* ssm_D   = (const float*)d_in[22];
  const float* ssm_wout= (const float*)d_in[23];
  float* out = (float*)d_out;

  char* ws = (char*)d_ws;
  size_t off = 0;
  auto alloc = [&](size_t bytes) -> char* {
    char* p = ws + off;
    off += (bytes + 255) & ~(size_t)255;
    return p;
  };

  h16* wqkv16 = (h16*)alloc(2L * DM * (3 * DM) * 2);   // fragment-packed
  h16* wo16   = (h16*)alloc(2L * DM * DM * 2);
  h16* w116   = (h16*)alloc(2L * DM * DFFN * 2);
  h16* w216   = (h16*)alloc(2L * DFFN * DM * 2);
  h16* embT   = (h16*)alloc((long)DM * VOCAB * 2);
  float* xbuf = (float*)alloc((long)WMAX * RPAD * DM * 4);
  h16* abuf   = (h16*)alloc((long)WMAX * RPAD * DM * 2);
  h16* qkvbuf = (h16*)alloc((long)WMAX * RPAD * (3 * DM) * 2);
  h16* obuf   = (h16*)alloc((long)WMAX * RPAD * DM * 2);
  h16* fbuf   = (h16*)alloc((long)WMAX * RPAD * DFFN * 2);
  h16* tlast  = (h16*)alloc((long)TROWS * DM * 2);
  float* wbuf = (float*)alloc((long)BATCH * TWIN * DM * 4);
  float* ssmst= (float*)alloc((2048L + 2L * BATCH * 1024 * 16) * 4);
  float* ssmrow = ssmst;            // [4][512]
  float* hstate = ssmst + 2048;     // [2][4][1024][16]

  // persistent-state init (determinism under graph replay) + weight packing
  zero_f32<<<64, 256, 0, stream>>>(ssmst, 2048L + 2L * BATCH * 1024 * 16);
  for (int l = 0; l < 2; ++l) {
    pack_b_f32<<<1024, 256, 0, stream>>>(wqkv + (long)l * DM * (3 * DM),
                                         wqkv16 + (long)l * DM * (3 * DM), DM, 3 * DM, 0);
    pack_b_f32<<<512, 256, 0, stream>>>(wo + (long)l * DM * DM,
                                        wo16 + (long)l * DM * DM, DM, DM, 0);
    pack_b_f32<<<1024, 256, 0, stream>>>(w1 + (long)l * DM * DFFN,
                                         w116 + (long)l * DM * DFFN, DM, DFFN, 0);
    pack_b_f32<<<1024, 256, 0, stream>>>(w2 + (long)l * DFFN * DM,
                                         w216 + (long)l * DFFN * DM, DFFN, DM, 0);
  }
  pack_b_f32<<<4096, 256, 0, stream>>>(emb, embT, DM, VOCAB, 1);  // tied lm-head

  for (int k = 0; k < 16; ++k) {
    int lo = (k == 0) ? 0 : 16 * k + 1;
    int hi = (k == 15) ? 255 : 16 * (k + 1);
    int W = (hi - lo + 1) * BATCH;          // 68 / 64 / 60 windows
    int M = W * RPAD;                       // multiple of 64 always
    int mb = M / 64;

    build_window<<<W, 256, 0, stream>>>(ids, emb, ssmrow, xbuf, lo);

    for (int l = 0; l < 2; ++l) {
      ln_rows<<<M, 128, 0, stream>>>(xbuf, abuf, ln1_g + l * DM, ln1_b + l * DM);
      gemm_wmma<0><<<dim3((3 * DM) / 256, mb), 256, 0, stream>>>(
          abuf, wqkv16 + (long)l * DM * (3 * DM), M, DM, 3 * DM, M,
          nullptr, nullptr, qkvbuf, 0, nullptr);
      attn_kernel<<<W * NHEAD, 256, 0, stream>>>(qkvbuf, obuf);
      gemm_wmma<2><<<dim3(DM / 256, mb), 256, 0, stream>>>(
          obuf, wo16 + (long)l * DM * DM, M, DM, DM, M,
          nullptr, xbuf, nullptr, 0, nullptr);
      ln_rows<<<M, 128, 0, stream>>>(xbuf, abuf, ln2_g + l * DM, ln2_b + l * DM);
      gemm_wmma<1><<<dim3(DFFN / 256, mb), 256, 0, stream>>>(
          abuf, w116 + (long)l * DM * DFFN, M, DM, DFFN, M,
          b1 + l * DFFN, nullptr, fbuf, 0, nullptr);
      gemm_wmma<2><<<dim3(DM / 256, mb), 256, 0, stream>>>(
          fbuf, w216 + (long)l * DFFN * DM, M, DFFN, DM, M,
          b2 + l * DM, xbuf, nullptr, 0, nullptr);
    }

    lnf_last<<<TROWS, 128, 0, stream>>>(xbuf, tlast, lnf_g, lnf_b, W);
    gemm_wmma<3><<<dim3(VOCAB / 256, TROWS / 64), 256, 0, stream>>>(
        tlast, embT, TROWS, DM, VOCAB, W,
        nullptr, nullptr, nullptr, lo, out);

    if (k < 15)
      pool_ssm_update<<<BATCH, 256, 0, stream>>>(
          ids, emb, ssmrow, hstate, wbuf, pool_q, pool_wk,
          ssm_win, ssm_wdt, ssm_bdt, ssm_Alog, ssm_wB, ssm_wC, ssm_D, ssm_wout, hi);
  }
}
